// GINE_4741643895755
// MI455X (gfx1250) — compile-verified, matrix operand order checked by reading, not run on hardware
//
#include <hip/hip_runtime.h>

#define WG 256
static constexpr int N_NODES = 50000;   // divisible by 16
static constexpr int N_EDGES = 800000;  // divisible by 64
static constexpr int C_IN    = 128;
static constexpr int C_HID   = 256;

typedef __attribute__((ext_vector_type(16))) __bf16 v16bf;
typedef __attribute__((ext_vector_type(8)))  float  v8f;

union BFrag { v16bf v; unsigned u[8]; };

__device__ __forceinline__ unsigned pack2(float x, float y) {
  union { __bf16 b[2]; unsigned u; } p;
  p.b[0] = (__bf16)x; p.b[1] = (__bf16)y;
  return p.u;
}

// A fragment (16x32 bf16 slice of a row-major LDS tile, row stride Kdw dwords).
// Per ISA layout: lane L holds row M (passed in); VGPR v holds K pair
// kbase = (v<4?0:16) + (L>=16?8:0) + (v&3)*2  (pair => one dword in LDS).
__device__ __forceinline__ v16bf load_a_frag(const unsigned* A, int m, int Kdw,
                                             int s, int lane) {
  const int half = lane >> 4;
  const int base = m * Kdw + s * 16 + half * 4;
  BFrag f;
#pragma unroll
  for (int v = 0; v < 8; ++v)
    f.u[v] = A[base + (v & 3) + ((v >> 2) << 3)];
  return f.v;
}

// B fragment from pre-packed weights: fragment for (col-tile t, k-step s) is
// 32 lanes x 16 contiguous bf16 (two global_load_b128 per lane).
__device__ __forceinline__ v16bf load_b_frag(const __bf16* pw, int t, int nK,
                                             int s, int lane) {
  const v16bf* p =
      (const v16bf*)(pw + ((size_t)(((t * nK) + s) * 32 + lane) << 4));
  return *p;
}

__device__ __forceinline__ void fatomic_add(float* addr, float v) {
  __hip_atomic_fetch_add(addr, v, __ATOMIC_RELAXED, __HIP_MEMORY_SCOPE_AGENT);
}

__device__ __forceinline__ v8f wmma_bf16(v16bf a, v16bf b, v8f c) {
  return __builtin_amdgcn_wmma_f32_16x16x32_bf16(false, a, false, b, (short)0,
                                                 c, false, false);
}

// ---- weight pre-pack: f32 row-major [K x D] -> bf16 WMMA-B fragment order ----
__global__ void pack_w_kernel(const float* __restrict__ W, __bf16* __restrict__ out,
                              int K, int D) {
  const int nK = K >> 5;
  const long long total = (long long)K * D;
  for (long long o = (long long)blockIdx.x * WG + threadIdx.x; o < total;
       o += (long long)gridDim.x * WG) {
    const int j    = (int)(o & 15);
    const int lane = (int)((o >> 4) & 31);
    const long long rest = o >> 9;
    const int s = (int)(rest % nK);
    const int t = (int)(rest / nK);
    const int v = j >> 1, half = j & 1;
    const int k = s * 32 + ((v >> 2) << 4) + ((lane >> 4) << 3) + ((v & 3) << 1) + half;
    const int n = (t << 4) + (lane & 15);
    out[o] = (__bf16)W[(size_t)k * D + n];
  }
}

__global__ void zero_kernel(float* __restrict__ p, long long n) {
  for (long long i = (long long)blockIdx.x * WG + threadIdx.x; i < n;
       i += (long long)gridDim.x * WG)
    p[i] = 0.f;
}

// ---- column sum / sumsq for BatchNorm (blockDim.x == C) ----
__global__ void col_stats_kernel(const float* __restrict__ x,
                                 float* __restrict__ stats, int N, int C) {
  const int c = threadIdx.x;
  const int r0 = blockIdx.x * 256;
  const int r1 = min(r0 + 256, N);
  float s = 0.f, s2 = 0.f;
  for (int r = r0; r < r1; ++r) {
    const float v = x[(size_t)r * C + c];
    s += v; s2 += v * v;
  }
  fatomic_add(&stats[c], s);
  fatomic_add(&stats[C + c], s2);
}

__global__ void bn_apply_kernel(const float* __restrict__ x,
                                const float* __restrict__ stats,
                                const float* __restrict__ g,
                                const float* __restrict__ b,
                                float* __restrict__ out, long long total, int C,
                                float invN) {
  for (long long i = (long long)blockIdx.x * WG + threadIdx.x; i < total;
       i += (long long)gridDim.x * WG) {
    const int c = (int)(i % C);
    const float mean = stats[c] * invN;
    const float var  = stats[C + c] * invN - mean * mean;
    const float sc   = g[c] * __frsqrt_rn(var + 1e-5f);
    out[i] = (x[i] - mean) * sc + b[c];
  }
}

// ---- fused edge layer: relu(h[src] + edge_attr @ eW + eb) scatter-added to agg
// Block = 64 edges (4 row tiles) x D cols; K = 128 (edge_attr dim) always.
// Each wave holds its 4 B K-step fragments in registers and reuses them
// across the 4 row tiles (16 WMMAs per B fetch). The row-tile loop is kept
// rolled (unroll 1) to bound register pressure; the 8 gather loads per tile
// are issued together so they clause and overlap.
__global__ void __launch_bounds__(WG, 2)
edge_msg_kernel(const float* __restrict__ h, const float* __restrict__ ea,
                const int* __restrict__ src, const int* __restrict__ dst,
                const __bf16* __restrict__ pw, const float* __restrict__ eb,
                float* __restrict__ agg, int D) {
  __shared__ unsigned ldsA[64 * 64];  // 64 edges x 128 bf16 (dword pairs) = 16KB
  __shared__ int sidx[64], didx[64];
  const int tid = threadIdx.x;
  const long long e0 = (long long)blockIdx.x * 64;
  if (tid < 64) {
    sidx[tid] = src[e0 + tid];
    didx[tid] = dst[e0 + tid];
  }
  const float2* ea2 = (const float2*)(ea + e0 * C_IN);
#pragma unroll
  for (int p = tid; p < 64 * 64; p += WG) {
    const float2 f = ea2[p];
    ldsA[p] = pack2(f.x, f.y);
  }
  __syncthreads();

  const int lane = tid & 31;
  const int wv = tid >> 5;
  const int tpw = D >> 7;             // column tiles per wave: 1 (D=128) or 2 (D=256)
  const int mlane = lane & 15;
  const int mbase = (lane >> 4) << 3;
#pragma unroll 1
  for (int tt = 0; tt < tpw; ++tt) {
    const int t = wv * tpw + tt;
    const v16bf b0 = load_b_frag(pw, t, 4, 0, lane);
    const v16bf b1 = load_b_frag(pw, t, 4, 1, lane);
    const v16bf b2 = load_b_frag(pw, t, 4, 2, lane);
    const v16bf b3 = load_b_frag(pw, t, 4, 3, lane);
    const int n = (t << 4) + mlane;
    const float bias = eb[n];
#pragma unroll 1
    for (int rt = 0; rt < 4; ++rt) {
      const int mrow = (rt << 4) + mlane;
      v8f acc = {0.f, 0.f, 0.f, 0.f, 0.f, 0.f, 0.f, 0.f};
      acc = wmma_bf16(load_a_frag(ldsA, mrow, 64, 0, lane), b0, acc);
      acc = wmma_bf16(load_a_frag(ldsA, mrow, 64, 1, lane), b1, acc);
      acc = wmma_bf16(load_a_frag(ldsA, mrow, 64, 2, lane), b2, acc);
      acc = wmma_bf16(load_a_frag(ldsA, mrow, 64, 3, lane), b3, acc);
      // batch the 8 gathers so they issue as one clause, then scatter
      unsigned doff[8];
      float hv[8];
#pragma unroll
      for (int r = 0; r < 8; ++r) {
        const int row = (rt << 4) + mbase + r;  // edge index within 64-edge tile
        hv[r]   = h[(size_t)((unsigned)sidx[row] * D + n)];
        doff[r] = (unsigned)didx[row] * D + n;
      }
#pragma unroll
      for (int r = 0; r < 8; ++r) {
        const float val = fmaxf(acc[r] + bias + hv[r], 0.f);
        fatomic_add(&agg[(size_t)doff[r]], val);
      }
    }
  }
}

// ---- node GEMM: out = tanh((h [+ agg]) @ W + b), 16 rows per block ----
__global__ void __launch_bounds__(WG, 2)
node_gemm_kernel(const float* __restrict__ h, const float* __restrict__ agg,
                 const __bf16* __restrict__ pw, const float* __restrict__ bias,
                 float* __restrict__ out, int K, int D) {
  extern __shared__ unsigned ldsA[];  // 16 x K bf16
  const int tid = threadIdx.x;
  const long long r0 = (long long)blockIdx.x * 16;
  const int Kdw = K >> 1;
  const float2* h2 = (const float2*)(h + r0 * K);
  const float2* a2 = agg ? (const float2*)(agg + r0 * K) : nullptr;
  const int tot = 16 * Kdw;
  for (int p = tid; p < tot; p += WG) {
    float2 f = h2[p];
    if (a2) {
      const float2 g = a2[p];
      f.x += g.x; f.y += g.y;
    }
    ldsA[p] = pack2(f.x, f.y);
  }
  __syncthreads();

  const int lane = tid & 31;
  const int wv = tid >> 5;
  const int nK = K >> 5;
  const int tpw = D >> 7;
  const int m = lane & 15;
  const int mbase = (lane >> 4) << 3;
#pragma unroll 1
  for (int tt = 0; tt < tpw; ++tt) {
    const int t = wv * tpw + tt;
    v8f acc = {0.f, 0.f, 0.f, 0.f, 0.f, 0.f, 0.f, 0.f};
    for (int s = 0; s < nK; ++s) {
      const v16bf a = load_a_frag(ldsA, m, Kdw, s, lane);
      const v16bf b = load_b_frag(pw, t, nK, s, lane);
      acc = wmma_bf16(a, b, acc);
    }
    const int n = (t << 4) + (lane & 15);
    const float bb = bias ? bias[n] : 0.f;
#pragma unroll
    for (int r = 0; r < 8; ++r)
      out[(r0 + mbase + r) * (size_t)D + n] = tanhf(acc[r] + bb);
  }
}

extern "C" void kernel_launch(void* const* d_in, const int* in_sizes, int n_in,
                              void* d_out, int out_size, void* d_ws, size_t ws_size,
                              hipStream_t stream) {
  (void)in_sizes; (void)n_in; (void)out_size; (void)ws_size;
  const float* x       = (const float*)d_in[0];
  const int*   eidx    = (const int*)d_in[1];   // JAX default: int32
  const float* ea      = (const float*)d_in[2];
  const float* bn_in_g = (const float*)d_in[3];
  const float* bn_in_b = (const float*)d_in[4];
  const float* e0_W    = (const float*)d_in[5];
  const float* e0_b    = (const float*)d_in[6];
  const float* n0_W    = (const float*)d_in[7];
  const float* n0_b    = (const float*)d_in[8];
  const float* bn0_g   = (const float*)d_in[9];
  const float* bn0_b   = (const float*)d_in[10];
  const float* e1_W    = (const float*)d_in[11];
  const float* e1_b    = (const float*)d_in[12];
  const float* n1_W    = (const float*)d_in[13];
  const float* n1_b    = (const float*)d_in[14];
  const float* bn1_g   = (const float*)d_in[15];
  const float* bn1_b   = (const float*)d_in[16];
  const float* fc_W    = (const float*)d_in[17];
  const int* src = eidx;
  const int* dst = eidx + N_EDGES;

  // workspace layout
  float* bufA  = (float*)d_ws;
  float* bufB  = bufA + (size_t)N_NODES * 256;
  float* agg   = bufB + (size_t)N_NODES * 256;
  float* stats = agg + (size_t)N_NODES * 256;
  __bf16* pw_e0 = (__bf16*)(stats + 1024);
  __bf16* pw_n0 = pw_e0 + 128 * 128;
  __bf16* pw_e1 = pw_n0 + 128 * 256;
  __bf16* pw_n1 = pw_e1 + 128 * 256;
  __bf16* pw_fc = pw_n1 + 256 * 256;

  // pre-pack weights into WMMA B-fragment order (bf16)
  pack_w_kernel<<<64,  WG, 0, stream>>>(e0_W, pw_e0, 128, 128);
  pack_w_kernel<<<128, WG, 0, stream>>>(n0_W, pw_n0, 128, 256);
  pack_w_kernel<<<128, WG, 0, stream>>>(e1_W, pw_e1, 128, 256);
  pack_w_kernel<<<256, WG, 0, stream>>>(n1_W, pw_n1, 256, 256);
  pack_w_kernel<<<128, WG, 0, stream>>>(fc_W, pw_fc, 256, 128);

  const int statBlocks = (N_NODES + 255) / 256;
  const float invN = 1.f / (float)N_NODES;

  // input BN: x -> bufA  (C=128)
  zero_kernel<<<2, WG, 0, stream>>>(stats, 512);
  col_stats_kernel<<<statBlocks, C_IN, 0, stream>>>(x, stats, N_NODES, C_IN);
  {
    const long long tot = (long long)N_NODES * C_IN;
    bn_apply_kernel<<<(int)((tot + WG - 1) / WG), WG, 0, stream>>>(
        x, stats, bn_in_g, bn_in_b, bufA, tot, C_IN, invN);
  }

  // ---- layer 0 ----
  zero_kernel<<<4096, WG, 0, stream>>>(agg, (long long)N_NODES * 128);
  edge_msg_kernel<<<N_EDGES / 64, WG, 0, stream>>>(bufA, ea, src, dst, pw_e0,
                                                   e0_b, agg, 128);
  node_gemm_kernel<<<N_NODES / 16, WG, 16 * 128 * 2, stream>>>(
      bufA, agg, pw_n0, n0_b, bufB, 128, 256);

  // BN0 in-place on bufB (C=256)
  zero_kernel<<<2, WG, 0, stream>>>(stats, 512);
  col_stats_kernel<<<statBlocks, C_HID, 0, stream>>>(bufB, stats, N_NODES, C_HID);
  {
    const long long tot = (long long)N_NODES * C_HID;
    bn_apply_kernel<<<(int)((tot + WG - 1) / WG), WG, 0, stream>>>(
        bufB, stats, bn0_g, bn0_b, bufB, tot, C_HID, invN);
  }

  // ---- layer 1 ----
  zero_kernel<<<8192, WG, 0, stream>>>(agg, (long long)N_NODES * 256);
  edge_msg_kernel<<<N_EDGES / 64, WG, 0, stream>>>(bufB, ea, src, dst, pw_e1,
                                                   e1_b, agg, 256);
  node_gemm_kernel<<<N_NODES / 16, WG, 16 * 256 * 2, stream>>>(
      bufB, agg, pw_n1, n1_b, bufA, 256, 256);

  // BN1 in-place on bufA (C=256)
  zero_kernel<<<2, WG, 0, stream>>>(stats, 512);
  col_stats_kernel<<<statBlocks, C_HID, 0, stream>>>(bufA, stats, N_NODES, C_HID);
  {
    const long long tot = (long long)N_NODES * C_HID;
    bn_apply_kernel<<<(int)((tot + WG - 1) / WG), WG, 0, stream>>>(
        bufA, stats, bn1_g, bn1_b, bufA, tot, C_HID, invN);
  }

  // ---- final FC: tanh(h @ fc_W) -> d_out ----
  node_gemm_kernel<<<N_NODES / 16, WG, 16 * 256 * 2, stream>>>(
      bufA, nullptr, pw_fc, nullptr, (float*)d_out, 256, 128);
}
